// RelationGNN_4088808866428
// MI455X (gfx1250) — matrix operand order repaired and checked by reading, not compiled, not run on hardware
//
#include <hip/hip_runtime.h>
#include <hip/hip_bf16.h>
#include <stdint.h>

#define N_NODES 100000
#define N_EDGES 500000
#define N_GRAPHS 64
#define IN_DIM 512
#define HID 256
#define N_REL 16

typedef __attribute__((ext_vector_type(16))) __bf16    v16bf;
typedef __attribute__((ext_vector_type(8)))  float     v8f;
typedef __attribute__((ext_vector_type(4)))  unsigned  u32x4;
typedef __attribute__((ext_vector_type(2)))  unsigned  u32x2;
typedef __attribute__((ext_vector_type(4)))  float     f32x4;
typedef __attribute__((ext_vector_type(4)))  int       i32x4;

// CDNA5 async global->LDS copy (ASYNCcnt-tracked), if the toolchain exposes it.
#if defined(__has_builtin)
#  if __has_builtin(__builtin_amdgcn_global_load_async_to_lds_b128) && \
      __has_builtin(__builtin_amdgcn_s_wait_asynccnt)
#    define USE_ASYNC_LDS 1
#  endif
#endif
#ifndef USE_ASYNC_LDS
#  define USE_ASYNC_LDS 0
#endif

__device__ __forceinline__ unsigned short f32_to_bf16(float f) {
  unsigned u = __builtin_bit_cast(unsigned, f);
  u += 0x7FFFu + ((u >> 16) & 1u);           // round-to-nearest-even
  return (unsigned short)(u >> 16);
}
__device__ __forceinline__ float bf16_to_f32(unsigned short h) {
  unsigned u = ((unsigned)h) << 16;
  return __builtin_bit_cast(float, u);
}

union Frag { v16bf v; u32x4 q[2]; };

// ---------------------------------------------------------------------------
// Weight repack: Wcat[K x 512] = [W_rel | W_root] -> bf16 in B-fragment order:
// out[(((kt*32 + nt)*32 + lane)*16) + i] = Wcat[kt*32 + (i/8)*16 + (lane/16)*8 + i%8][nt*16 + lane%16]
// ---------------------------------------------------------------------------
__global__ void pack_wcat(const float* __restrict__ Wrel, const float* __restrict__ Wroot,
                          unsigned short* __restrict__ out, int K) {
  int t = blockIdx.x * blockDim.x + threadIdx.x;
  if (t >= K * 512) return;
  int i    = t & 15;
  int lane = (t >> 4) & 31;
  int nt   = (t >> 9) & 31;
  int kt   = t >> 14;
  int k = kt * 32 + (i >> 3) * 16 + (lane >> 4) * 8 + (i & 7);
  int n = nt * 16 + (lane & 15);
  float v = (n < HID) ? Wrel[k * HID + n] : Wroot[k * HID + (n - HID)];
  out[t] = f32_to_bf16(v);
}

// W_edge[512 x 16] -> bf16 fragment order (single N-tile)
__global__ void pack_edge(const float* __restrict__ We, unsigned short* __restrict__ out) {
  int t = blockIdx.x * blockDim.x + threadIdx.x;   // 512*16 = 8192
  if (t >= 2 * HID * N_REL) return;
  int i    = t & 15;
  int lane = (t >> 4) & 31;
  int kt   = t >> 9;
  int k = kt * 32 + (i >> 3) * 16 + (lane >> 4) * 8 + (i & 7);
  int n = lane & 15;
  out[t] = f32_to_bf16(We[k * N_REL + n]);
}

// ---------------------------------------------------------------------------
// GEMM: X[32-row strip, KDIM] @ Wcat[KDIM x 512] (bf16 WMMA, f32 accum).
// 8 waves/block; each wave owns 64 cols x 32 rows = 2x4 fragments
// (8 WMMAs per k-step against 4 ds_load_b128 + 8 global_load_b128).
// cols [0,256)  -> Hrel (bf16, relation term), cols [256,512) -> agg (f32 root term).
// ---------------------------------------------------------------------------
template<int KDIM, bool IN_F32>
__global__ __launch_bounds__(256)
void gemm_layer(const void* __restrict__ Xin, const unsigned short* __restrict__ Wpack,
                unsigned short* __restrict__ Hrel, float* __restrict__ agg) {
  constexpr int BM   = 32;
  constexpr int LSTR = KDIM + 8;                 // bf16 elems; keeps 16B alignment
  __shared__ __align__(16) unsigned short atile[BM * LSTR];
  const int row0 = blockIdx.x * BM;
  const int tid  = threadIdx.x;

  if constexpr (IN_F32) {
    const float* X = (const float*)Xin;
    for (int idx = tid; idx < BM * KDIM / 4; idx += 256) {
      int r  = idx / (KDIM / 4);
      int c4 = idx % (KDIM / 4);
      f32x4 f = *(const f32x4*)(X + (size_t)(row0 + r) * KDIM + c4 * 4);
      u32x2 o;
      o[0] = (unsigned)f32_to_bf16(f[0]) | ((unsigned)f32_to_bf16(f[1]) << 16);
      o[1] = (unsigned)f32_to_bf16(f[2]) | ((unsigned)f32_to_bf16(f[3]) << 16);
      *(u32x2*)&atile[r * LSTR + c4 * 4] = o;
    }
  } else {
    const unsigned short* X = (const unsigned short*)Xin;
#if USE_ASYNC_LDS
    for (int idx = tid; idx < BM * KDIM / 8; idx += 256) {
      int r  = idx / (KDIM / 8);
      int c8 = idx % (KDIM / 8);
      __builtin_amdgcn_global_load_async_to_lds_b128(
          (__attribute__((address_space(1))) i32x4*)(X + (size_t)(row0 + r) * KDIM + c8 * 8),
          (__attribute__((address_space(3))) i32x4*)&atile[r * LSTR + c8 * 8],
          0, 0);
    }
    __builtin_amdgcn_s_wait_asynccnt(0);
#else
    for (int idx = tid; idx < BM * KDIM / 8; idx += 256) {
      int r  = idx / (KDIM / 8);
      int c8 = idx % (KDIM / 8);
      u32x4 q = *(const u32x4*)(X + (size_t)(row0 + r) * KDIM + c8 * 8);
      *(u32x4*)&atile[r * LSTR + c8 * 8] = q;
    }
#endif
  }
  __syncthreads();

  const int wave = tid >> 5;          // 0..7 -> col base wave*64
  const int lane = tid & 31;
  const int half = lane >> 4;         // K-half selector for A/B fragments
  const int m    = lane & 15;         // M row (A) / N col (B,D)

  v8f acc[2][4] = {};
  for (int kt = 0; kt < KDIM / 32; ++kt) {
    Frag a0, a1;
    const unsigned short* ar0 = &atile[m * LSTR + kt * 32];
    const unsigned short* ar1 = &atile[(16 + m) * LSTR + kt * 32];
    a0.q[0] = *(const u32x4*)(ar0 + half * 8);
    a0.q[1] = *(const u32x4*)(ar0 + 16 + half * 8);
    a1.q[0] = *(const u32x4*)(ar1 + half * 8);
    a1.q[1] = *(const u32x4*)(ar1 + 16 + half * 8);
    const unsigned short* bbase =
        Wpack + (((size_t)kt * 32 + (size_t)wave * 4) * 32 + lane) * 16;
    Frag b[4];
#pragma unroll
    for (int n = 0; n < 4; ++n) {
      b[n].q[0] = *(const u32x4*)(bbase + n * 512);
      b[n].q[1] = *(const u32x4*)(bbase + n * 512 + 8);
    }
#pragma unroll
    for (int n = 0; n < 4; ++n) {
      acc[0][n] = __builtin_amdgcn_wmma_f32_16x16x32_bf16(false, a0.v, false, b[n].v, (short)0, acc[0][n], false, false);
      acc[1][n] = __builtin_amdgcn_wmma_f32_16x16x32_bf16(false, a1.v, false, b[n].v, (short)0, acc[1][n], false, false);
    }
  }

  const int colbase = wave * 64;
#pragma unroll
  for (int mi = 0; mi < 2; ++mi) {
#pragma unroll
    for (int ntl = 0; ntl < 4; ++ntl) {
      int col = colbase + ntl * 16 + m;
#pragma unroll
      for (int j = 0; j < 8; ++j) {
        int row = row0 + mi * 16 + half * 8 + j;
        float v = acc[mi][ntl][j];
        if (col < HID) Hrel[(size_t)row * HID + col] = f32_to_bf16(v);
        else           agg [(size_t)row * HID + (col - HID)] = v;
      }
    }
  }
}

// ---------------------------------------------------------------------------
// Edge scatter: agg[dst] += Hrel[src]  (agg pre-initialized with root term).
// 8 edges per block (1 per wave); each lane: one b128 gather + 8 f32 atomics.
// ---------------------------------------------------------------------------
__global__ __launch_bounds__(256)
void scatter_edges(const unsigned short* __restrict__ Hrel, const int* __restrict__ eidx,
                   float* __restrict__ agg) {
  int e    = blockIdx.x * 8 + (threadIdx.x >> 5);
  int lane = threadIdx.x & 31;
  int s = eidx[e];
  int d = eidx[N_EDGES + e];
  int c = lane * 8;
  u32x4 q = *(const u32x4*)(Hrel + (size_t)s * HID + c);
  float* dst = agg + (size_t)d * HID + c;
#pragma unroll
  for (int k = 0; k < 4; ++k) {
    atomicAdd(dst + 2 * k,     bf16_to_f32((unsigned short)(q[k] & 0xFFFFu)));
    atomicAdd(dst + 2 * k + 1, bf16_to_f32((unsigned short)(q[k] >> 16)));
  }
}

// relu(agg + bias) -> bf16 activations, 8 elems/thread, b128 in/out
__global__ __launch_bounds__(256)
void combine_relu(const float* __restrict__ agg, const float* __restrict__ bias,
                  unsigned short* __restrict__ xout) {
  size_t t = ((size_t)blockIdx.x * 256 + threadIdx.x) * 8;
  int col = (int)(t & (size_t)(HID - 1));
  f32x4 a0 = *(const f32x4*)(agg + t);
  f32x4 a1 = *(const f32x4*)(agg + t + 4);
  f32x4 b0 = *(const f32x4*)(bias + col);
  f32x4 b1 = *(const f32x4*)(bias + col + 4);
  unsigned short p[8];
#pragma unroll
  for (int j = 0; j < 4; ++j) {
    float v0 = a0[j] + b0[j]; v0 = v0 > 0.f ? v0 : 0.f;
    float v1 = a1[j] + b1[j]; v1 = v1 > 0.f ? v1 : 0.f;
    p[j]     = f32_to_bf16(v0);
    p[j + 4] = f32_to_bf16(v1);
  }
  u32x4 o;
  o[0] = (unsigned)p[0] | ((unsigned)p[1] << 16);
  o[1] = (unsigned)p[2] | ((unsigned)p[3] << 16);
  o[2] = (unsigned)p[4] | ((unsigned)p[5] << 16);
  o[3] = (unsigned)p[6] | ((unsigned)p[7] << 16);
  *(u32x4*)(xout + t) = o;
}

// ---------------------------------------------------------------------------
// Edge head: 1 wave handles 32 edges. A = gathered [x[src]|x[dst]] 32x512 bf16
// in LDS; 2 WMMAs per k-step (B fragment reused) -> 32x16 logits (+bias).
// ---------------------------------------------------------------------------
__global__ __launch_bounds__(32)
void edge_head(const unsigned short* __restrict__ xbuf, const int* __restrict__ eidx,
               const unsigned short* __restrict__ WpackE, const float* __restrict__ b_edge,
               float* __restrict__ logits) {
  constexpr int LSTR = 2 * HID + 8;  // 520
  __shared__ __align__(16) unsigned short atile[32 * LSTR];
  const int ebase = blockIdx.x * 32;
  const int tid = threadIdx.x;
#pragma unroll
  for (int t = tid; t < 64; t += 32) {
    int r    = t >> 1;               // edge row in tile
    int hsel = t & 1;                // 0 = src half, 1 = dst half
    int e    = ebase + r;
    int node = hsel ? eidx[N_EDGES + e] : eidx[e];
    const u32x4* srcp = (const u32x4*)(xbuf + (size_t)node * HID);
    u32x4* dstp = (u32x4*)&atile[r * LSTR + hsel * HID];
#pragma unroll
    for (int q = 0; q < HID / 8; ++q) dstp[q] = srcp[q];
  }
  __syncthreads();

  const int lane = tid;
  const int half = lane >> 4;
  const int m    = lane & 15;
  v8f acc0 = {}, acc1 = {};
#pragma unroll
  for (int kt = 0; kt < (2 * HID) / 32; ++kt) {
    Frag a0, a1, b;
    const unsigned short* ar0 = &atile[m * LSTR + kt * 32];
    const unsigned short* ar1 = &atile[(16 + m) * LSTR + kt * 32];
    a0.q[0] = *(const u32x4*)(ar0 + half * 8);
    a0.q[1] = *(const u32x4*)(ar0 + 16 + half * 8);
    a1.q[0] = *(const u32x4*)(ar1 + half * 8);
    a1.q[1] = *(const u32x4*)(ar1 + 16 + half * 8);
    const unsigned short* bbase = WpackE + ((size_t)kt * 32 + lane) * 16;
    b.q[0] = *(const u32x4*)(bbase);
    b.q[1] = *(const u32x4*)(bbase + 8);
    acc0 = __builtin_amdgcn_wmma_f32_16x16x32_bf16(false, a0.v, false, b.v, (short)0, acc0, false, false);
    acc1 = __builtin_amdgcn_wmma_f32_16x16x32_bf16(false, a1.v, false, b.v, (short)0, acc1, false, false);
  }
  const float bb = b_edge[m];
#pragma unroll
  for (int j = 0; j < 8; ++j) {
    int e0 = ebase + half * 8 + j;
    logits[(size_t)e0 * N_REL + m] = acc0[j] + bb;
    int e1 = ebase + 16 + half * 8 + j;
    logits[(size_t)e1 * N_REL + m] = acc1[j] + bb;
  }
}

// ---------------------------------------------------------------------------
// Global mean pool
// ---------------------------------------------------------------------------
__global__ void pool_init(float* __restrict__ buf, int n) {
  int t = blockIdx.x * blockDim.x + threadIdx.x;
  if (t < n) buf[t] = 0.f;
}
__global__ __launch_bounds__(256)
void pool_scatter(const unsigned short* __restrict__ xbuf, const int* __restrict__ batch,
                  float* __restrict__ sums, float* __restrict__ counts) {
  int n    = blockIdx.x * 8 + (threadIdx.x >> 5);
  int lane = threadIdx.x & 31;
  int g = batch[n];
  int c = lane * 8;
  u32x4 q = *(const u32x4*)(xbuf + (size_t)n * HID + c);
  float* dst = sums + (size_t)g * HID + c;
#pragma unroll
  for (int k = 0; k < 4; ++k) {
    atomicAdd(dst + 2 * k,     bf16_to_f32((unsigned short)(q[k] & 0xFFFFu)));
    atomicAdd(dst + 2 * k + 1, bf16_to_f32((unsigned short)(q[k] >> 16)));
  }
  if (lane == 0) atomicAdd(&counts[g], 1.0f);
}
__global__ void pool_final(const float* __restrict__ sums, const float* __restrict__ counts,
                           float* __restrict__ out) {
  int t = blockIdx.x * blockDim.x + threadIdx.x;   // 64*256
  int g = t >> 8;
  out[t] = sums[t] / fmaxf(counts[g], 1.0f);
}

// ---------------------------------------------------------------------------
extern "C" void kernel_launch(void* const* d_in, const int* in_sizes, int n_in,
                              void* d_out, int out_size, void* d_ws, size_t ws_size,
                              hipStream_t stream) {
  const float* node_feats = (const float*)d_in[0];
  const int*   eidx       = (const int*)d_in[1];
  const int*   batch      = (const int*)d_in[2];
  const float* Wrel0 = (const float*)d_in[3];
  const float* Wroot0= (const float*)d_in[4];
  const float* b0    = (const float*)d_in[5];
  const float* Wrel1 = (const float*)d_in[6];
  const float* Wroot1= (const float*)d_in[7];
  const float* b1    = (const float*)d_in[8];
  const float* Wrel2 = (const float*)d_in[9];
  const float* Wroot2= (const float*)d_in[10];
  const float* b2    = (const float*)d_in[11];
  const float* Wedge = (const float*)d_in[12];
  const float* bedge = (const float*)d_in[13];
  float* out = (float*)d_out;

  char* ws = (char*)d_ws;
  size_t off = 0;
  auto alloc = [&](size_t bytes) -> char* {
    char* p = ws + off;
    off = (off + bytes + 255) & ~(size_t)255;
    return p;
  };
  unsigned short* Wp0  = (unsigned short*)alloc((size_t)IN_DIM * 512 * 2);
  unsigned short* Wp1  = (unsigned short*)alloc((size_t)HID * 512 * 2);
  unsigned short* Wp2  = (unsigned short*)alloc((size_t)HID * 512 * 2);
  unsigned short* WpE  = (unsigned short*)alloc((size_t)2 * HID * N_REL * 2);
  unsigned short* Hrel = (unsigned short*)alloc((size_t)N_NODES * HID * 2);
  float*          agg  = (float*)alloc((size_t)N_NODES * HID * 4);
  unsigned short* xbuf = (unsigned short*)alloc((size_t)N_NODES * HID * 2);
  float*          sums = (float*)alloc((size_t)(N_GRAPHS * HID + N_GRAPHS) * 4);
  float*          counts = sums + N_GRAPHS * HID;

  // weight repack (tiny)
  pack_wcat<<<(IN_DIM * 512) / 256, 256, 0, stream>>>(Wrel0, Wroot0, Wp0, IN_DIM);
  pack_wcat<<<(HID * 512) / 256, 256, 0, stream>>>(Wrel1, Wroot1, Wp1, HID);
  pack_wcat<<<(HID * 512) / 256, 256, 0, stream>>>(Wrel2, Wroot2, Wp2, HID);
  pack_edge<<<(2 * HID * N_REL + 255) / 256, 256, 0, stream>>>(Wedge, WpE);

  // layer 0 (K=512, f32 input)
  gemm_layer<IN_DIM, true><<<N_NODES / 32, 256, 0, stream>>>(node_feats, Wp0, Hrel, agg);
  scatter_edges<<<N_EDGES / 8, 256, 0, stream>>>(Hrel, eidx, agg);
  combine_relu<<<(N_NODES * HID) / (256 * 8), 256, 0, stream>>>(agg, b0, xbuf);
  // layer 1 (K=256, bf16 input)
  gemm_layer<HID, false><<<N_NODES / 32, 256, 0, stream>>>(xbuf, Wp1, Hrel, agg);
  scatter_edges<<<N_EDGES / 8, 256, 0, stream>>>(Hrel, eidx, agg);
  combine_relu<<<(N_NODES * HID) / (256 * 8), 256, 0, stream>>>(agg, b1, xbuf);
  // layer 2
  gemm_layer<HID, false><<<N_NODES / 32, 256, 0, stream>>>(xbuf, Wp2, Hrel, agg);
  scatter_edges<<<N_EDGES / 8, 256, 0, stream>>>(Hrel, eidx, agg);
  combine_relu<<<(N_NODES * HID) / (256 * 8), 256, 0, stream>>>(agg, b2, xbuf);

  // edge head -> out[0 .. N_EDGES*16)
  edge_head<<<N_EDGES / 32, 32, 0, stream>>>(xbuf, eidx, WpE, bedge, out);

  // mean pool -> out[N_EDGES*16 ..)
  int pn = N_GRAPHS * HID + N_GRAPHS;
  pool_init<<<(pn + 255) / 256, 256, 0, stream>>>(sums, pn);
  pool_scatter<<<N_NODES / 8, 256, 0, stream>>>(xbuf, batch, sums, counts);
  pool_final<<<(N_GRAPHS * HID) / 256, 256, 0, stream>>>(sums, counts, out + (size_t)N_EDGES * N_REL);
}
